// BahdanauAttention_27565100106226
// MI455X (gfx1250) — matrix operand order repaired and checked by reading, not compiled, not run on hardware
//
#include <hip/hip_runtime.h>
#include <hip/hip_bf16.h>
#include <math.h>

// Problem sizes (fixed by the reference)
#define BB 32
#define TT 2048
#define HH 1024
#define EE 1024
#define TM 64              // t-rows per workgroup in the main kernel
#define RT (TM / 16)       // 4 row-tiles per k-tile
#define KT (HH / 16)       // 64 k-tiles of 16 columns
#define KT_PER_WAVE (KT / 8)

typedef __attribute__((ext_vector_type(16))) __bf16 v16bf;
typedef __attribute__((ext_vector_type(8)))  __bf16 v8bf;
typedef __attribute__((ext_vector_type(8)))  float  v8f;

__device__ __forceinline__ unsigned short f2bf_rne(float x) {
    unsigned int u = __float_as_uint(x);
    unsigned int r = u + 0x7FFFu + ((u >> 16) & 1u);
    return (unsigned short)(r >> 16);
}

__device__ __forceinline__ float fast_tanh(float x) {
#if __has_builtin(__builtin_amdgcn_tanhf)
    return __builtin_amdgcn_tanhf(x);   // native V_TANH_F32 on gfx1250
#else
    return tanhf(x);
#endif
}

// ---------------------------------------------------------------------------
// 0) zero the atomic-accumulated buffers (scores workspace + context output)
// ---------------------------------------------------------------------------
__global__ void bahdanau_init_kernel(float* __restrict__ scores, int ns,
                                     float* __restrict__ ctx, int nc) {
    int i = blockIdx.x * 256 + threadIdx.x;
    if (i < ns) scores[i] = 0.0f;
    if (i < nc) ctx[i] = 0.0f;
}

// ---------------------------------------------------------------------------
// 1) dec_proj[b,k] = sum_h dec_hidden[b,h] * W1[k,h]   (tiny, VALU is fine)
// ---------------------------------------------------------------------------
__global__ __launch_bounds__(256)
void bahdanau_decproj_kernel(const float* __restrict__ dh,
                             const float* __restrict__ W1,
                             float* __restrict__ dp) {
    int t = blockIdx.x * 256 + threadIdx.x;      // 0 .. B*H-1
    int b = t >> 10;
    int k = t & (HH - 1);
    const float* __restrict__ x = dh + (size_t)b * HH;
    const float* __restrict__ w = W1 + (size_t)k * HH;
    float a0 = 0.f, a1 = 0.f, a2 = 0.f, a3 = 0.f;
    for (int h = 0; h < HH; h += 4) {
        a0 = fmaf(x[h + 0], w[h + 0], a0);
        a1 = fmaf(x[h + 1], w[h + 1], a1);
        a2 = fmaf(x[h + 2], w[h + 2], a2);
        a3 = fmaf(x[h + 3], w[h + 3], a3);
    }
    dp[t] = (a0 + a1) + (a2 + a3);
}

// ---------------------------------------------------------------------------
// 2) Pack W2 (f32 [H,E]) into bf16 B-operand fragments, fragment-major:
//    w2f[((kt*32 + ec)*32 + lane)*16 + i] = bf16( W2[kt*16 + (lane&15)]
//                                                   [ec*32 + (lane<16?0:16) + i] )
//    B-matrix (32x16, 16-bit) k order used here, same as A: lanes 0-15 hold K=0..15 (N=lane),
//    lanes 16-31 hold K=16..31 (N=lane-16).
// ---------------------------------------------------------------------------
__global__ __launch_bounds__(256)
void bahdanau_w2pack_kernel(const float* __restrict__ W2,
                            unsigned short* __restrict__ w2f) {
    int fid = blockIdx.x * 256 + threadIdx.x;    // 0 .. 1048575
    int i    = fid & 15;
    int lane = (fid >> 4) & 31;
    int ec   = (fid >> 9) & 31;
    int kt   = fid >> 14;
    int k = kt * 16 + (lane & 15);
    int e = ec * 32 + ((lane >> 4) ? 16 : 0) + i;
    w2f[fid] = f2bf_rne(W2[(size_t)k * EE + e]);
}

// ---------------------------------------------------------------------------
// 3) Main fused kernel: enc_proj GEMM (bf16 WMMA) + tanh + dot(v) -> scores
//    One workgroup = one (b, 64-row t-tile), 128 KB dynamic LDS.
//    8 waves split the 64 k-tiles; each B fragment feeds 4 WMMAs.
// ---------------------------------------------------------------------------
__global__ __launch_bounds__(256)
void bahdanau_scores_kernel(const float* __restrict__ enc,          // [B,T,E]
                            const unsigned short* __restrict__ w2f, // fragments
                            const float* __restrict__ dp,           // [B,H]
                            const float* __restrict__ vvec,         // [H]
                            float* __restrict__ scores) {           // [B,T]
    extern __shared__ unsigned short sA[];   // TM*EE bf16 = 128 KB, row-major

    const int wg    = blockIdx.x;            // 0 .. B*(T/TM)-1
    const int b     = wg / (TT / TM);
    const int tbase = (wg % (TT / TM)) * TM;
    const int tid   = threadIdx.x;
    const int wave  = tid >> 5;
    const int lane  = tid & 31;
    const int col   = lane & 15;             // N (k col) and M (t row) selector
    const int halfhi = lane >> 4;            // 0: lanes 0-15, 1: lanes 16-31
    const int kstart = halfhi ? 16 : 0;      // A-operand K sub-block start

    // ---- stage enc[b, tbase..tbase+63, :] as bf16 into LDS (contiguous blk)
    const float2* __restrict__ src =
        (const float2*)(enc + ((size_t)b * TT + tbase) * EE);
    unsigned int* __restrict__ dst = (unsigned int*)sA;
    for (int i = tid; i < (TM * EE) / 2; i += 256) {
        float2 f = src[i];
        dst[i] = (unsigned int)f2bf_rne(f.x) | ((unsigned int)f2bf_rne(f.y) << 16);
    }
    __syncthreads();

    float rowacc[RT][8];
    #pragma unroll
    for (int rt = 0; rt < RT; ++rt)
        #pragma unroll
        for (int r = 0; r < 8; ++r) rowacc[rt][r] = 0.0f;

    for (int kk = 0; kk < KT_PER_WAVE; ++kk) {
        const int kt = wave * KT_PER_WAVE + kk;
        const int kglob = kt * 16 + col;
        const float dpk = dp[(size_t)b * HH + kglob];
        const float vk  = vvec[kglob];
        const unsigned short* __restrict__ w2kt = w2f + (size_t)kt * 32 * 512;

        v8f acc[RT];
        #pragma unroll
        for (int rt = 0; rt < RT; ++rt) acc[rt] = (v8f){};

        #pragma unroll 2
        for (int ec = 0; ec < EE / 32; ++ec) {
            // B fragment: 32 contiguous bytes per lane, coalesced across wave
            v16bf bfrag = *(const v16bf*)(w2kt + ec * 512 + lane * 16);
            #pragma unroll
            for (int rt = 0; rt < RT; ++rt) {
                // A fragment, same contiguous k order as B: two 16B runs from LDS
                const unsigned short* ap =
                    sA + (rt * 16 + col) * EE + ec * 32 + kstart;
                v8bf alo = *(const v8bf*)(ap);
                v8bf ahi = *(const v8bf*)(ap + 8);
                v16bf afrag = __builtin_shufflevector(alo, ahi,
                    0,1,2,3,4,5,6,7,8,9,10,11,12,13,14,15);
                acc[rt] = __builtin_amdgcn_wmma_f32_16x16x32_bf16(
                    false, afrag, false, bfrag, (short)0, acc[rt], false, false);
            }
        }

        // epilogue: tanh(dec_proj + enc_proj) * v[k], reduce over 16 k-cols
        #pragma unroll
        for (int rt = 0; rt < RT; ++rt) {
            #pragma unroll
            for (int r = 0; r < 8; ++r) {
                float c = fast_tanh(acc[rt][r] + dpk) * vk;
                c += __shfl_xor(c, 1, 16);
                c += __shfl_xor(c, 2, 16);
                c += __shfl_xor(c, 4, 16);
                c += __shfl_xor(c, 8, 16);
                rowacc[rt][r] += c;          // row = rt*16 + 8*halfhi + r
            }
        }
    }

    // lanes 0 and 16 carry the distinct rows of each half
    if (col == 0) {
        float* __restrict__ sb = scores + (size_t)b * TT + tbase + 8 * halfhi;
        #pragma unroll
        for (int rt = 0; rt < RT; ++rt)
            #pragma unroll
            for (int r = 0; r < 8; ++r)
                atomicAdd(&sb[rt * 16 + r], rowacc[rt][r]);
    }
}

// ---------------------------------------------------------------------------
// 4) softmax over T per batch row
// ---------------------------------------------------------------------------
__global__ __launch_bounds__(256)
void bahdanau_softmax_kernel(const float* __restrict__ scores,
                             float* __restrict__ attn) {
    __shared__ float sh[TT];
    __shared__ float sred[9];
    const int b = blockIdx.x, tid = threadIdx.x;
    const int wave = tid >> 5, lane = tid & 31;
    const float* __restrict__ sc = scores + (size_t)b * TT;

    float m = -3.4e38f;
    for (int t = tid; t < TT; t += 256) { float x = sc[t]; sh[t] = x; m = fmaxf(m, x); }
    for (int o = 16; o; o >>= 1) m = fmaxf(m, __shfl_xor(m, o, 32));
    if (!lane) sred[wave] = m;
    __syncthreads();
    if (tid == 0) { float mm = sred[0]; for (int i = 1; i < 8; ++i) mm = fmaxf(mm, sred[i]); sred[8] = mm; }
    __syncthreads();
    m = sred[8];

    float s = 0.f;
    for (int t = tid; t < TT; t += 256) { float ev = __expf(sh[t] - m); sh[t] = ev; s += ev; }
    for (int o = 16; o; o >>= 1) s += __shfl_xor(s, o, 32);
    __syncthreads();
    if (!lane) sred[wave] = s;
    __syncthreads();
    if (tid == 0) { float ss = sred[0]; for (int i = 1; i < 8; ++i) ss += sred[i]; sred[8] = ss; }
    __syncthreads();
    const float inv = 1.0f / sred[8];

    float* __restrict__ ao = attn + (size_t)b * TT;
    for (int t = tid; t < TT; t += 256) ao[t] = sh[t] * inv;
}

// ---------------------------------------------------------------------------
// 5) context[b,e] = sum_t attn[b,t] * enc[b,t,e]  (t-split + f32 atomics)
// ---------------------------------------------------------------------------
#define TSPLIT 8
__global__ __launch_bounds__(256)
void bahdanau_context_kernel(const float* __restrict__ enc,
                             const float* __restrict__ attn,
                             float* __restrict__ ctx) {
    const int bid = blockIdx.x;                  // B * (E/256) * TSPLIT blocks
    const int ts = bid & (TSPLIT - 1);
    const int ech = (bid >> 3) & ((EE / 256) - 1);
    const int b = bid >> 5;
    const int e = ech * 256 + threadIdx.x;
    const int t0 = ts * (TT / TSPLIT);
    const float* __restrict__ at = attn + (size_t)b * TT + t0;
    const float* __restrict__ ep = enc + ((size_t)b * TT + t0) * EE + e;
    float acc = 0.f;
    for (int t = 0; t < TT / TSPLIT; ++t)
        acc = fmaf(at[t], ep[(size_t)t * EE], acc);
    atomicAdd(&ctx[(size_t)b * EE + e], acc);
}

// ---------------------------------------------------------------------------
extern "C" void kernel_launch(void* const* d_in, const int* in_sizes, int n_in,
                              void* d_out, int out_size, void* d_ws, size_t ws_size,
                              hipStream_t stream) {
    const float* dec_hidden = (const float*)d_in[0];   // [B,H]
    const float* enc        = (const float*)d_in[1];   // [B,T,E]
    const float* W1         = (const float*)d_in[2];   // [H,H]
    const float* W2         = (const float*)d_in[3];   // [H,E]
    const float* vvec       = (const float*)d_in[4];   // [H]

    float* out_ctx  = (float*)d_out;                   // [B,E]
    float* out_attn = (float*)d_out + (size_t)BB * EE; // [B,T]

    char* ws = (char*)d_ws;
    float* dp               = (float*)ws;                          // B*H f32
    float* scores           = (float*)(ws + 131072);               // B*T f32
    unsigned short* w2f     = (unsigned short*)(ws + 131072 + 262144); // 2 MB

    bahdanau_init_kernel<<<(BB * TT + 255) / 256, 256, 0, stream>>>(
        scores, BB * TT, out_ctx, BB * EE);
    bahdanau_decproj_kernel<<<(BB * HH) / 256, 256, 0, stream>>>(dec_hidden, W1, dp);
    bahdanau_w2pack_kernel<<<(KT * 32 * 32 * 16) / 256, 256, 0, stream>>>(W2, w2f);
    bahdanau_scores_kernel<<<BB * (TT / TM), 256, TM * EE * 2, stream>>>(
        enc, w2f, dp, vvec, scores);
    bahdanau_softmax_kernel<<<BB, 256, 0, stream>>>(scores, out_attn);
    bahdanau_context_kernel<<<BB * (EE / 256) * TSPLIT, 256, 0, stream>>>(
        enc, out_attn, out_ctx);
}
